// MoEMlp_27685359190687
// MI455X (gfx1250) — matrix operand order repaired and checked by reading, not compiled, not run on hardware
//
#include <hip/hip_runtime.h>
#include <hip/hip_bf16.h>
#include <math.h>

// Problem constants: x [4,4096,1024] f32, two experts 1024->4096->1024.
#define C_DIM 1024
#define H_DIM 4096
#define M_TOK 16384

// GEMM tiling: block tile 128(M) x 128(N), K chunks of 64, double-buffered LDS.
// 8 waves in 4(M) x 2(N); each wave owns 32x64 = 2x4 WMMA tiles.
#define BM 128
#define BN 128
#define BK 64
#define APAD 72   // LDS row stride (halfs); 144B = multiple of 16B
#define BPAD 72

typedef __attribute__((ext_vector_type(16))) __bf16 v16bf;
typedef __attribute__((ext_vector_type(8)))  __bf16 v8bf;
typedef __attribute__((ext_vector_type(8)))  float  v8f;

union F16x16 { v16bf v; v8bf h[2]; };

static __device__ __forceinline__ unsigned short f2bf(float f) {
    unsigned int u = __builtin_bit_cast(unsigned int, f);
    u += 0x7FFFu + ((u >> 16) & 1u);   // round-to-nearest-even
    return (unsigned short)(u >> 16);
}

// Low 32 bits of a generic pointer to __shared__ = addrspace(3) LDS byte offset.
static __device__ __forceinline__ unsigned ldsOff(const void* p) {
    return (unsigned)(unsigned long long)(uintptr_t)p;
}

// Async global->LDS, 16B per lane. INST_OFFSET applies to BOTH lds and global
// addresses, so one lds-offset VGPR + one 64b address pair serves 4 copies.
#define ASYNC_CP16(ldsoff, gptr, OFF)                                        \
    asm volatile("global_load_async_to_lds_b128 %0, %1, off offset:" #OFF    \
                 :: "v"(ldsoff), "v"(gptr) : "memory")
#define STAGE64B(ldsoff, src)                                                \
    do { ASYNC_CP16(ldsoff, src, 0);  ASYNC_CP16(ldsoff, src, 16);           \
         ASYNC_CP16(ldsoff, src, 32); ASYNC_CP16(ldsoff, src, 48); } while (0)
#define ASYNC_WAIT_N(N) asm volatile("s_wait_asynccnt " #N ::: "memory")

// ---------------------------------------------------------------------------
// Token dispatch: compact token indices per expert.
// ---------------------------------------------------------------------------
__global__ void zero_counts_kernel(int* counts) {
    if (threadIdx.x < 2) counts[threadIdx.x] = 0;
}

__global__ void build_index_kernel(const long long* __restrict__ tt,
                                   int* __restrict__ counts,
                                   int* __restrict__ idxbuf) {
    int i = blockIdx.x * blockDim.x + threadIdx.x;
    if (i < M_TOK) {
        long long t = tt[i];
        if (t == 0 || t == 1) {
            int e = (int)t;
            int p = atomicAdd(&counts[e], 1);
            idxbuf[e * M_TOK + p] = i;
        }
    }
}

// ---------------------------------------------------------------------------
// x f32 -> bf16 (one pass; 8 elems/thread).
// ---------------------------------------------------------------------------
__global__ __launch_bounds__(256) void cvt_x_kernel(
        const float* __restrict__ src, unsigned short* __restrict__ dst) {
    size_t i = ((size_t)blockIdx.x * 256 + threadIdx.x) * 8;
    float4 a = *(const float4*)(src + i);
    float4 b = *(const float4*)(src + i + 4);
    unsigned int p0 = (unsigned)f2bf(a.x) | ((unsigned)f2bf(a.y) << 16);
    unsigned int p1 = (unsigned)f2bf(a.z) | ((unsigned)f2bf(a.w) << 16);
    unsigned int p2 = (unsigned)f2bf(b.x) | ((unsigned)f2bf(b.y) << 16);
    unsigned int p3 = (unsigned)f2bf(b.z) | ((unsigned)f2bf(b.w) << 16);
    *(uint4*)(dst + i) = make_uint4(p0, p1, p2, p3);
}

// ---------------------------------------------------------------------------
// Weight conversion f32 [K][N] -> bf16 transposed [N][K] (one-time, LDS tiled).
// ---------------------------------------------------------------------------
__global__ __launch_bounds__(256) void transpose_cvt_kernel(
        const float* __restrict__ src, unsigned short* __restrict__ dst,
        int K, int N) {
    __shared__ float tile[32][33];
    int kb = blockIdx.y * 32, nb = blockIdx.x * 32;
    int tx = threadIdx.x, ty = threadIdx.y;   // 32 x 8
    #pragma unroll
    for (int i = 0; i < 32; i += 8)
        tile[ty + i][tx] = src[(size_t)(kb + ty + i) * N + nb + tx];
    __syncthreads();
    #pragma unroll
    for (int i = 0; i < 32; i += 8)
        dst[(size_t)(nb + ty + i) * K + kb + tx] = f2bf(tile[tx][ty + i]);
}

// ---------------------------------------------------------------------------
// Shared WMMA inner compute: 2x4 tiles per wave from LDS buffers.
// ---------------------------------------------------------------------------
static __device__ __forceinline__ void wmma_compute(
        const unsigned short* Ab, const unsigned short* Bb,
        int wm, int wn, int lrow, int hi, v8f acc[2][4]) {
    #pragma unroll
    for (int kk = 0; kk < BK; kk += 32) {
        F16x16 a[2], b[4];
        int aklo = kk + hi * 8;   // A: lanes 0-15 K{0..7,16..23}; 16-31 K{8..15,24..31}
        int bklo = kk + hi * 8;   // B: lanes 0-15 K{0..7,16..23}; 16-31 K{8..15,24..31}
        #pragma unroll
        for (int t = 0; t < 2; ++t) {
            int ar = wm * 32 + t * 16 + lrow;
            a[t].h[0] = *(const v8bf*)(Ab + ar * APAD + aklo);
            a[t].h[1] = *(const v8bf*)(Ab + ar * APAD + aklo + 16);
        }
        #pragma unroll
        for (int t = 0; t < 4; ++t) {
            int bc = wn * 64 + t * 16 + lrow;
            b[t].h[0] = *(const v8bf*)(Bb + bc * BPAD + bklo);
            b[t].h[1] = *(const v8bf*)(Bb + bc * BPAD + bklo + 16);
        }
        #pragma unroll
        for (int tm = 0; tm < 2; ++tm)
            #pragma unroll
            for (int tn = 0; tn < 4; ++tn)
                acc[tm][tn] = __builtin_amdgcn_wmma_f32_16x16x32_bf16(
                    false, a[tm].v, false, b[tn].v,
                    (short)0, acc[tm][tn], false, false);
    }
}

// ---------------------------------------------------------------------------
// GEMM1: h = gelu(gather(xbf) @ w1_e + b1_e) -> compacted bf16 h [M][H]
// Fully async-staged, double-buffered.
// ---------------------------------------------------------------------------
__global__ __launch_bounds__(256) void gemm1_gelu_kernel(
        const unsigned short* __restrict__ xbf,   // [M][C] bf16
        const unsigned short* __restrict__ w1T,   // [2][H][C] bf16 (transposed)
        const float* __restrict__ b1_s, const float* __restrict__ b1_l,
        const int* __restrict__ counts, const int* __restrict__ idxbuf,
        unsigned short* __restrict__ hbuf) {
    int e   = blockIdx.z;
    int cnt = counts[e];
    int m0  = blockIdx.y * BM;
    if (m0 >= cnt) return;                       // uniform early exit
    int base = e ? counts[0] : 0;
    int n0  = blockIdx.x * BN;
    const unsigned short* wT = w1T + (size_t)e * H_DIM * C_DIM;
    const float* bias = e ? b1_l : b1_s;
    const int* idxE = idxbuf + e * M_TOK;

    __shared__ unsigned short As[2 * BM * APAD];
    __shared__ unsigned short Bs[2 * BN * BPAD];

    int tid  = threadIdx.x;
    int lane = tid & 31;
    int wv   = tid >> 5;
    int wm   = wv >> 1, wn = wv & 1;
    int lrow = lane & 15;
    int hi   = lane >> 4;

    v8f acc[2][4] = {};

    // Staging geometry: 2 threads/row, 64B (32 halfs) each.
    int srow = tid >> 1;
    int scol = (tid & 1) * 32;
    unsigned asO[2] = { ldsOff(&As[srow * APAD + scol]),
                        ldsOff(&As[BM * APAD + srow * APAD + scol]) };
    unsigned bsO[2] = { ldsOff(&Bs[srow * BPAD + scol]),
                        ldsOff(&Bs[BN * BPAD + srow * BPAD + scol]) };

    // Gather index hoisted out of the K loop; async DMA does the gather.
    int p   = m0 + srow;
    int tok = (p < cnt) ? idxE[p] : 0;
    const unsigned short* aRow = xbf + (size_t)tok * C_DIM + scol;
    const unsigned short* bRow = wT + (size_t)(n0 + srow) * C_DIM + scol;

    // Prologue: chunk 0 -> buffer 0
    STAGE64B(asO[0], aRow);
    STAGE64B(bsO[0], bRow);

    const int NIT = C_DIM / BK;
    for (int it = 0; it < NIT; ++it) {
        int cur = it & 1;
        if (it + 1 < NIT) {                      // prefetch next chunk
            const unsigned short* na = aRow + (it + 1) * BK;
            const unsigned short* nb = bRow + (it + 1) * BK;
            STAGE64B(asO[cur ^ 1], na);
            STAGE64B(bsO[cur ^ 1], nb);
            ASYNC_WAIT_N(8);                     // current chunk complete (in-order)
        } else {
            ASYNC_WAIT_N(0);
        }
        __syncthreads();
        wmma_compute(As + cur * BM * APAD, Bs + cur * BN * BPAD,
                     wm, wn, lrow, hi, acc);
        __syncthreads();                         // protect cur buffer before reuse
    }

    // Epilogue: bias + exact GELU, store compacted bf16 h.
    #pragma unroll
    for (int tm = 0; tm < 2; ++tm) {
        #pragma unroll
        for (int tn = 0; tn < 4; ++tn) {
            int ng = n0 + wn * 64 + tn * 16 + lrow;
            float bv = bias[ng];
            #pragma unroll
            for (int r = 0; r < 8; ++r) {
                int ml = wm * 32 + tm * 16 + hi * 8 + r;
                int pp = m0 + ml;
                if (pp < cnt) {
                    float v = acc[tm][tn][r] + bv;
                    float g = 0.5f * v * (1.0f + erff(v * 0.70710678118654752f));
                    hbuf[(size_t)(base + pp) * H_DIM + ng] = f2bf(g);
                }
            }
        }
    }
}

// ---------------------------------------------------------------------------
// GEMM2: out = h @ w2_e + b2_e, scattered back to original token rows (f32).
// ---------------------------------------------------------------------------
__global__ __launch_bounds__(256) void gemm2_out_kernel(
        const unsigned short* __restrict__ hbuf,  // [M][H] bf16 compacted
        const unsigned short* __restrict__ w2T,   // [2][C][H] bf16 (transposed)
        const float* __restrict__ b2_s, const float* __restrict__ b2_l,
        const int* __restrict__ counts, const int* __restrict__ idxbuf,
        float* __restrict__ out) {
    int e   = blockIdx.z;
    int cnt = counts[e];
    int m0  = blockIdx.y * BM;
    if (m0 >= cnt) return;
    int base = e ? counts[0] : 0;
    int n0  = blockIdx.x * BN;
    const unsigned short* wT = w2T + (size_t)e * C_DIM * H_DIM;
    const float* bias = e ? b2_l : b2_s;
    const int* idxE = idxbuf + e * M_TOK;

    __shared__ unsigned short As[2 * BM * APAD];
    __shared__ unsigned short Bs[2 * BN * BPAD];

    int tid  = threadIdx.x;
    int lane = tid & 31;
    int wv   = tid >> 5;
    int wm   = wv >> 1, wn = wv & 1;
    int lrow = lane & 15;
    int hi   = lane >> 4;

    v8f acc[2][4] = {};

    int srow = tid >> 1;
    int scol = (tid & 1) * 32;
    unsigned asO[2] = { ldsOff(&As[srow * APAD + scol]),
                        ldsOff(&As[BM * APAD + srow * APAD + scol]) };
    unsigned bsO[2] = { ldsOff(&Bs[srow * BPAD + scol]),
                        ldsOff(&Bs[BN * BPAD + srow * BPAD + scol]) };

    int pA  = m0 + srow;
    int rpA = base + ((pA < cnt) ? pA : 0);      // clamp at expert boundary
    const unsigned short* aRow = hbuf + (size_t)rpA * H_DIM + scol;
    const unsigned short* bRow = wT + (size_t)(n0 + srow) * H_DIM + scol;

    STAGE64B(asO[0], aRow);
    STAGE64B(bsO[0], bRow);

    const int NIT = H_DIM / BK;
    for (int it = 0; it < NIT; ++it) {
        int cur = it & 1;
        if (it + 1 < NIT) {
            const unsigned short* na = aRow + (it + 1) * BK;
            const unsigned short* nb = bRow + (it + 1) * BK;
            STAGE64B(asO[cur ^ 1], na);
            STAGE64B(bsO[cur ^ 1], nb);
            ASYNC_WAIT_N(8);
        } else {
            ASYNC_WAIT_N(0);
        }
        __syncthreads();
        wmma_compute(As + cur * BM * APAD, Bs + cur * BN * BPAD,
                     wm, wn, lrow, hi, acc);
        __syncthreads();
    }

    // Epilogue: bias, scatter f32 to original token rows.
    #pragma unroll
    for (int tm = 0; tm < 2; ++tm) {
        #pragma unroll
        for (int tn = 0; tn < 4; ++tn) {
            int ng = n0 + wn * 64 + tn * 16 + lrow;
            float bv = bias[ng];
            #pragma unroll
            for (int r = 0; r < 8; ++r) {
                int ml = wm * 32 + tm * 16 + hi * 8 + r;
                int pp = m0 + ml;
                if (pp < cnt) {
                    int tokO = idxE[pp];
                    out[(size_t)tokO * C_DIM + ng] = acc[tm][tn][r] + bv;
                }
            }
        }
    }
}

// ---------------------------------------------------------------------------
// Host launch
// ---------------------------------------------------------------------------
extern "C" void kernel_launch(void* const* d_in, const int* in_sizes, int n_in,
                              void* d_out, int out_size, void* d_ws, size_t ws_size,
                              hipStream_t stream) {
    (void)in_sizes; (void)n_in; (void)out_size; (void)ws_size;

    const float*     x    = (const float*)d_in[0];
    const long long* tt   = (const long long*)d_in[1];
    const float*     w1_s = (const float*)d_in[2];
    const float*     b1_s = (const float*)d_in[3];
    const float*     w2_s = (const float*)d_in[4];
    const float*     b2_s = (const float*)d_in[5];
    const float*     w1_l = (const float*)d_in[6];
    const float*     b1_l = (const float*)d_in[7];
    const float*     w2_l = (const float*)d_in[8];
    const float*     b2_l = (const float*)d_in[9];
    float* out = (float*)d_out;

    // Workspace (~202 MB):
    //  [counts:2i][idx 2*M i][xbf M*C][w1T 2*H*C][w2T 2*C*H][h M*H], all bf16 tails
    char* ws = (char*)d_ws;
    int* counts = (int*)ws;
    int* idxbuf = (int*)(ws + 1024);
    unsigned short* xbf  = (unsigned short*)(ws + 1024 + (size_t)2 * M_TOK * sizeof(int));
    unsigned short* w1T  = xbf + (size_t)M_TOK * C_DIM;
    unsigned short* w2T  = w1T + (size_t)2 * H_DIM * C_DIM;
    unsigned short* hbuf = w2T + (size_t)2 * C_DIM * H_DIM;

    zero_counts_kernel<<<1, 64, 0, stream>>>(counts);
    build_index_kernel<<<M_TOK / 256, 256, 0, stream>>>(tt, counts, idxbuf);

    cvt_x_kernel<<<(M_TOK * (size_t)C_DIM) / (256 * 8), 256, 0, stream>>>(x, xbf);

    dim3 tb(32, 8);
    transpose_cvt_kernel<<<dim3(H_DIM / 32, C_DIM / 32), tb, 0, stream>>>(
        w1_s, w1T, C_DIM, H_DIM);
    transpose_cvt_kernel<<<dim3(H_DIM / 32, C_DIM / 32), tb, 0, stream>>>(
        w1_l, w1T + (size_t)H_DIM * C_DIM, C_DIM, H_DIM);
    transpose_cvt_kernel<<<dim3(C_DIM / 32, H_DIM / 32), tb, 0, stream>>>(
        w2_s, w2T, H_DIM, C_DIM);
    transpose_cvt_kernel<<<dim3(C_DIM / 32, H_DIM / 32), tb, 0, stream>>>(
        w2_l, w2T + (size_t)C_DIM * H_DIM, H_DIM, C_DIM);

    gemm1_gelu_kernel<<<dim3(H_DIM / BN, M_TOK / BM, 2), 256, 0, stream>>>(
        xbf, w1T, b1_s, b1_l, counts, idxbuf, hbuf);
    gemm2_out_kernel<<<dim3(C_DIM / BN, M_TOK / BM, 2), 256, 0, stream>>>(
        hbuf, w2T, b2_s, b2_l, counts, idxbuf, out);
}
